// FilteringActLayer_79817672229331
// MI455X (gfx1250) — matrix operand order repaired and checked by reading, not compiled, not run on hardware
//
#include <hip/hip_runtime.h>

// Fused StyleGAN3-style filtered activation for MI455X (gfx1250, wave32).
// bias -> up2x separable FIR (polyphase) -> leaky*gain+clamp -> down2x separable FIR.
// All FIRs are banded 16x16x32 matmuls on V_WMMA_F32_16X16X32_F16 (f32 accumulate).
// Band matrices live in block-shared LDS (sparse-filled once); all LDS traffic is
// contiguous (packed f16-pair stores, vectorized fragment loads).

typedef __attribute__((ext_vector_type(16))) _Float16 v16h;
typedef __attribute__((ext_vector_type(8)))  float    v8f;
typedef __attribute__((ext_vector_type(2)))  _Float16 h2;
typedef __attribute__((ext_vector_type(4)))  _Float16 v4h;

#define LDK   36                 // row stride in halfs (72B: 8B-aligned rows)
#define PLANE (32 * LDK)         // 1152 halfs
#define ZHALF (16 * LDK)         // 576 halfs (one 16-row Z plane)
#define WAVES_PER_BLOCK 4
#define WAVE_SMEM (6 * PLANE)    // 6 planes per wave (with aliasing)
#define BAND_ROWS 96             // upE[0,32) upO[32,64) dnE[64,80) dnO[80,96)

// A fragment (16x32 f16) from row-major [m][k] tile.
// Wave32 layout (ISA 7.12.2): lane<16 -> K 0..7/16..23, lane>=16 -> K 8..15/24..31.
__device__ __forceinline__ v16h data_A(int lane, const _Float16* base, int ld) {
  v16h a;
  const int m  = lane & 15;
  const int kb = (lane & 16) ? 8 : 0;
#pragma unroll
  for (int h = 0; h < 16; ++h) {
    const int k = ((h & 8) << 1) + kb + (h & 7);
    a[h] = base[m * ld + k];
  }
  return a;
}

// B fragment (32x16 f16) from [n][k]-major tile: lane<16 -> K 0..7/16..23, lane>=16 -> K 8..15/24..31.
__device__ __forceinline__ v16h data_B(int lane, const _Float16* base, int ld) {
  v16h bf;
  const int n  = lane & 15;
  const int kb = (lane & 16) ? 8 : 0;
#pragma unroll
  for (int h = 0; h < 16; ++h) bf[h] = base[n * ld + ((h & 8) << 1) + kb + (h & 7)];
  return bf;
}

__device__ __forceinline__ v8f wmma16(v16h a, v16h b, v8f c) {
  return __builtin_amdgcn_wmma_f32_16x16x32_f16(false, a, false, b, (short)0, c,
                                                false, false);
}

// D fragment (M=r+mb, N=n) stored [N][M]: contiguous in r -> packed f16-pair stores.
__device__ __forceinline__ void store_D_packed(int lane, _Float16* base, int ld, v8f d) {
  const int n  = lane & 15;
  const int mb = (lane & 16) ? 8 : 0;
  _Float16* p = base + n * ld + mb;
#pragma unroll
  for (int r = 0; r < 8; r += 2) {
    h2 v;
    v.x = (_Float16)d[r];
    v.y = (_Float16)d[r + 1];
    *(h2*)(p + r) = v;
  }
}

__device__ __forceinline__ void store_D_packed_act(int lane, _Float16* base, int ld, v8f d,
                                                   float slope, float gain, float cl) {
  const int n  = lane & 15;
  const int mb = (lane & 16) ? 8 : 0;
  _Float16* p = base + n * ld + mb;
#pragma unroll
  for (int r = 0; r < 8; r += 2) {
    float a = d[r], bv = d[r + 1];
    a  = (a  >= 0.0f ? a  : a  * slope) * gain;
    bv = (bv >= 0.0f ? bv : bv * slope) * gain;
    a  = fminf(fmaxf(a,  -cl), cl);
    bv = fminf(fmaxf(bv, -cl), cl);
    h2 v;
    v.x = (_Float16)a;
    v.y = (_Float16)bv;
    *(h2*)(p + r) = v;
  }
}

__global__ __launch_bounds__(32 * WAVES_PER_BLOCK, 1)
void FilteringActLayer_79817672229331_kernel(
    const float* __restrict__ x, const float* __restrict__ b,
    const float* __restrict__ fu, const float* __restrict__ fd,
    const float* __restrict__ gainp, const float* __restrict__ slopep,
    const float* __restrict__ clampp, float* __restrict__ out) {
  __shared__ _Float16 sband[BAND_ROWS * LDK];                 // 6912 B, block-shared
  __shared__ _Float16 smem[WAVES_PER_BLOCK][WAVE_SMEM];       // 55296 B

  const int lane  = threadIdx.x & 31;
  const int wave  = threadIdx.x >> 5;
  const int tile  = blockIdx.x * WAVES_PER_BLOCK + wave;  // 65536 tiles
  const int plane = tile >> 6;                            // n*C + c
  const int t     = tile & 63;
  const int h0    = (t >> 3) << 4;
  const int w0    = (t & 7) << 4;
  const int c     = plane & 127;

  const float gain = gainp[0], slope = slopep[0], cl = clampp[0];
  // Polyphase taps: up-phases carry the UP-gain (2x per pass); down split by parity.
  float ge[6], go[6], de[6], dm[6];
#pragma unroll
  for (int s = 0; s < 6; ++s) {
    ge[s] = 2.0f * fu[2 * s];
    go[s] = 2.0f * fu[2 * s + 1];
    dm[s] = fd[2 * s];       // pairs with odd up-phase
    de[s] = fd[2 * s + 1];   // pairs with even up-phase
  }

  // ---- Shared band tables (once per block): tab[row][k] = taps[row+5-k].
  //      Sparse fill: zero the row, then scatter the 6 non-zero taps. ----
  {
    const int row = threadIdx.x;
    if (row < BAND_ROWS) {
      _Float16* dst = sband + row * LDK;
      const int r = (row < 64) ? (row & 31) : (row & 15);
#pragma unroll
      for (int k = 0; k < 32; k += 4) *(v4h*)(dst + k) = (v4h)0;  // 8B chunks, aligned
      float cf[6];
#pragma unroll
      for (int s = 0; s < 6; ++s) {
        float v = ge[s];
        v = (row >= 32) ? go[s] : v;
        v = (row >= 64) ? de[s] : v;
        v = (row >= 80) ? dm[s] : v;
        cf[s] = v;
      }
#pragma unroll
      for (int s = 0; s < 6; ++s) {
        const int k = r + 5 - s;
        if (k < 32) dst[k] = (_Float16)cf[s];
      }
    }
  }
  const _Float16* ubE = sband;             // up even-phase band, rows m/mw
  const _Float16* ubO = sband + 32 * LDK;  // up odd-phase band
  const _Float16* dbE = sband + 64 * LDK;  // down band (fd odd taps), rows h/w
  const _Float16* dbO = sband + 80 * LDK;  // down band (fd even taps)

  const float bias = b[c];
  const float* xp = x + (size_t)plane * (128 * 128);
  float*       op = out + (size_t)plane * (128 * 128);

  // Per-wave planes with aliasing (all stages per-wave, LDS in-order per wave):
  //   P0: xT [j][i]      -> later Y00 [mw][mh]
  //   P1: yH_e [m][j]    -> later Z planes [h][mw] (2 x 16 rows)
  //   P2: yH_o [m][j]
  //   P3..P5: Y01, Y10, Y11  [mw][mh]
  _Float16* S  = smem[wave];
  _Float16* P0 = S;
  _Float16* P1 = S + 1 * PLANE;
  _Float16* P2 = S + 2 * PLANE;
  _Float16* P3 = S + 3 * PLANE;
  _Float16* P4 = S + 4 * PLANE;
  _Float16* P5 = S + 5 * PLANE;
  _Float16* Yp[2][2] = {{P0, P3}, {P4, P5}};

  // ---- Stage 0: load halo [h0-5,h0+26]x[w0-5,w0+26], +bias, zero-pad, f16, [j][i] ----
  {
    const int i0 = h0 - 5, j0 = w0 - 5;
    const int j = j0 + lane;
    const bool jok = (j >= 0) && (j < 128);
#pragma unroll
    for (int r = 0; r < 32; r += 2) {
      const int ia = i0 + r, ib = i0 + r + 1;
      const float va = (jok && ia >= 0 && ia < 128) ? xp[ia * 128 + j] + bias : 0.0f;
      const float vb = (jok && ib >= 0 && ib < 128) ? xp[ib * 128 + j] + bias : 0.0f;
      h2 v;
      v.x = (_Float16)va;
      v.y = (_Float16)vb;
      *(h2*)(P0 + lane * LDK + r) = v;
    }
  }
  __syncthreads();  // band tables are cross-wave

  // ---- Stage 1: UpH (transposed).  yH_p stored [m][j].
  //      D[j][m] = sum_i xT[j][i] * upband_p[m][i]   (A = xT data, B = band) ----
#pragma unroll
  for (int p = 0; p < 2; ++p) {
    _Float16* yh = p ? P2 : P1;
    const _Float16* ub = p ? ubO : ubE;
#pragma unroll
    for (int mt = 0; mt < 2; ++mt) {                       // j-tile (M)
      const v16h A = data_A(lane, P0 + mt * 16 * LDK, LDK);
#pragma unroll
      for (int nt = 0; nt < 2; ++nt) {                     // m-tile (N)
        const v16h B = data_B(lane, ub + nt * 16 * LDK, LDK);
        v8f acc = {};
        acc = wmma16(A, B, acc);
        store_D_packed(lane, yh + nt * 16 * LDK + mt * 16, LDK, acc);
      }
    }
  }
  asm volatile("s_wait_dscnt 0" ::: "memory");

  // ---- Stage 2: UpW + act.  Y_pq stored [mw][mh].
  //      D[mh][mw] = sum_j yH_p[mh][j] * upband_q[mw][j]  (A = yH data, B = band) ----
#pragma unroll
  for (int p = 0; p < 2; ++p) {
    const _Float16* yh = p ? P2 : P1;
#pragma unroll
    for (int q = 0; q < 2; ++q) {
      const _Float16* ub = q ? ubO : ubE;
      _Float16* Y = Yp[p][q];
#pragma unroll
      for (int mt = 0; mt < 2; ++mt) {                     // mh-tile (M)
        const v16h A = data_A(lane, yh + mt * 16 * LDK, LDK);
#pragma unroll
        for (int nt = 0; nt < 2; ++nt) {                   // mw-tile (N)
          const v16h B = data_B(lane, ub + nt * 16 * LDK, LDK);
          v8f acc = {};
          acc = wmma16(A, B, acc);
          store_D_packed_act(lane, Y + nt * 16 * LDK + mt * 16, LDK, acc,
                             slope, gain, cl);
        }
      }
    }
  }
  asm volatile("s_wait_dscnt 0" ::: "memory");

  // ---- Stage 3: DownH.  Z_q stored [h][mw] (reuses P1).
  //      D[mw][h] = sum_mh Y_eq[mw-major] * dbE[h][mh] + Y_oq * dbO[h][mh] ----
#pragma unroll
  for (int q = 0; q < 2; ++q) {
    _Float16* Z = P1 + q * ZHALF;
#pragma unroll
    for (int mt = 0; mt < 2; ++mt) {                       // mw-tile (M)
      const v16h A0 = data_A(lane, Yp[0][q] + mt * 16 * LDK, LDK);
      const v16h Be = data_B(lane, dbE, LDK);
      v8f acc = {};
      acc = wmma16(A0, Be, acc);
      const v16h A1 = data_A(lane, Yp[1][q] + mt * 16 * LDK, LDK);
      const v16h Bo = data_B(lane, dbO, LDK);
      acc = wmma16(A1, Bo, acc);
      store_D_packed(lane, Z + mt * 16, LDK, acc);         // n = h (16 rows)
    }
  }
  asm volatile("s_wait_dscnt 0" ::: "memory");

  // ---- Stage 4: DownW + store.  out[h][w] = sum_mw Z_e[h][mw]*dbE[w][mw] + Z_o*dbO ----
  {
    const v16h A0 = data_A(lane, P1, LDK);                 // Z_e rows h, k = mw
    const v16h Be = data_B(lane, dbE, LDK);
    v8f acc = {};
    acc = wmma16(A0, Be, acc);
    const v16h A1 = data_A(lane, P1 + ZHALF, LDK);
    const v16h Bo = data_B(lane, dbO, LDK);
    acc = wmma16(A1, Bo, acc);
    const int n  = lane & 15;
    const int mb = (lane & 16) ? 8 : 0;
#pragma unroll
    for (int r = 0; r < 8; ++r) {
      op[(h0 + r + mb) * 128 + (w0 + n)] = acc[r];
    }
  }
}

extern "C" void kernel_launch(void* const* d_in, const int* in_sizes, int n_in,
                              void* d_out, int out_size, void* d_ws, size_t ws_size,
                              hipStream_t stream) {
  (void)in_sizes; (void)n_in; (void)out_size; (void)d_ws; (void)ws_size;
  const float* x  = (const float*)d_in[0];
  const float* b  = (const float*)d_in[1];
  const float* fu = (const float*)d_in[2];
  const float* fd = (const float*)d_in[3];
  const float* g  = (const float*)d_in[4];
  const float* s  = (const float*)d_in[5];
  const float* cl = (const float*)d_in[6];
  float* out = (float*)d_out;

  const int total_tiles = 8 * 128 * 64;  // N*C planes x (128/16)^2 tiles = 65536
  dim3 grid(total_tiles / WAVES_PER_BLOCK);
  dim3 block(32 * WAVES_PER_BLOCK);
  hipLaunchKernelGGL(FilteringActLayer_79817672229331_kernel, grid, block, 0, stream,
                     x, b, fu, fd, g, s, cl, out);
}
